// Encoder_47493748359494
// MI455X (gfx1250) — matrix operand order repaired and checked by reading, not compiled, not run on hardware
//
#include <hip/hip_runtime.h>

#define B_DIM 64
#define T_DIM 512
#define E_DIM 1024
#define H_DIM 1024
#define H3    (3 * H_DIM)
#define MTOT  (B_DIM * T_DIM)   // 32768 rows for the input GEMM
#define SCAN_BLOCKS 32

typedef __bf16 bf16_t;
typedef __attribute__((ext_vector_type(16))) __bf16 v16bf;
typedef __attribute__((ext_vector_type(8)))  __bf16 v8bf;
typedef __attribute__((ext_vector_type(8)))  float   v8f;

// ---------------------------------------------------------------------------
// WMMA fragment helpers (wave32, bf16 16x16x32, contiguous K order, same for A and B)
// A is [M,K] row-major bf16; B is stored transposed as BT = [N,K] row-major
// so each lane's fragment is a contiguous run of K values.
// ---------------------------------------------------------------------------
__device__ __forceinline__ v16bf load_frag_a(const bf16_t* __restrict__ A,
                                             int row0, int k0, int ld, int lane) {
  const int half = (lane >> 4) & 1;      // lanes 16-31 take K 16..31
  const int m    = lane & 15;
  const bf16_t* p = A + (size_t)(row0 + m) * ld + k0 + half * 16;
  v8bf lo = *(const v8bf*)p;             // K = base .. base+7
  v8bf hi = *(const v8bf*)(p + 8);       // K = base+8 .. base+15
  v16bf r;
#pragma unroll
  for (int i = 0; i < 8; ++i) { r[i] = lo[i]; r[i + 8] = hi[i]; }
  return r;
}

__device__ __forceinline__ v16bf load_frag_b(const bf16_t* __restrict__ BT,
                                             int ncol0, int k0, int ld, int lane) {
  const int half = (lane >> 4) & 1;      // lanes 16-31 take K 16..31
  const int n    = lane & 15;
  const bf16_t* p = BT + (size_t)(ncol0 + n) * ld + k0 + half * 16;
  return *(const v16bf*)p;               // 16 contiguous K values (32B)
}

__device__ __forceinline__ v8f wmma_bf16(v16bf a, v16bf b, v8f c) {
  return __builtin_amdgcn_wmma_f32_16x16x32_bf16(
      /*neg_a=*/false, a, /*neg_b=*/false, b,
      /*c_mod=*/(short)0, c, /*reuse_a=*/false, /*reuse_b=*/false);
}

// ---------------------------------------------------------------------------
// One-time prep kernels
// ---------------------------------------------------------------------------
__global__ void init_state(const float* __restrict__ hidden,  // [2,B,H]
                           float* __restrict__ state_out,     // [2,B,H] (in d_out)
                           bf16_t* __restrict__ hbf0,
                           bf16_t* __restrict__ hbf1,
                           unsigned int* __restrict__ cnt) {
  int i = blockIdx.x * blockDim.x + threadIdx.x;
  if (i == 0) { cnt[0] = 0u; cnt[1] = 0u; }
  if (i < 2 * B_DIM * H_DIM) {
    float v = hidden[i];
    state_out[i] = v;
    if (i < B_DIM * H_DIM) hbf0[i] = (bf16_t)v;
    else                   hbf1[i - B_DIM * H_DIM] = (bf16_t)v;
  }
}

__global__ void f32_to_bf16(const float* __restrict__ src,
                            bf16_t* __restrict__ dst, int n) {
  int i = (blockIdx.x * blockDim.x + threadIdx.x) * 4;
  if (i + 3 < n) {
    float4 v = *(const float4*)(src + i);
    dst[i + 0] = (bf16_t)v.x; dst[i + 1] = (bf16_t)v.y;
    dst[i + 2] = (bf16_t)v.z; dst[i + 3] = (bf16_t)v.w;
  }
}

// src [K][N] fp32  ->  dst [N][K] bf16   (weights pre-transposed for WMMA B)
__global__ void transpose_f32_to_bf16(const float* __restrict__ src,
                                      bf16_t* __restrict__ dst, int K, int N) {
  __shared__ bf16_t tile[32][33];
  int n0 = blockIdx.x * 32, k0 = blockIdx.y * 32;
  for (int i = threadIdx.y; i < 32; i += 8)
    tile[i][threadIdx.x] = (bf16_t)src[(size_t)(k0 + i) * N + n0 + threadIdx.x];
  __syncthreads();
  for (int i = threadIdx.y; i < 32; i += 8)
    dst[(size_t)(n0 + i) * K + k0 + threadIdx.x] = tile[threadIdx.x][i];
}

// ---------------------------------------------------------------------------
// Input GEMM:  C[M,N] = A[M,K](bf16) * BT[N,K]^T(bf16) + bias   (f32 accum)
// Block: 256 threads = 8 waves; wave tile 32x64 (2 M-tiles x 4 N-tiles).
// Block tile 64(M) x 256(N). M%64==0, N%256==0, K%32==0 guaranteed here.
// ---------------------------------------------------------------------------
__global__ __launch_bounds__(256) void gemm_bf16_tn(
    const bf16_t* __restrict__ A, const bf16_t* __restrict__ BT,
    const float* __restrict__ bias, float* __restrict__ C,
    int M, int N, int K) {
  const int lane  = threadIdx.x & 31;
  const int wave  = threadIdx.x >> 5;
  const int wm    = wave & 1;
  const int wn    = wave >> 1;
  const int mbase = blockIdx.y * 64 + wm * 32;
  const int nbase = blockIdx.x * 256 + wn * 64;
  const int half  = (lane >> 4) & 1;
  const int nlane = lane & 15;
  (void)M;

  v8f acc[2][4];
#pragma unroll
  for (int nt = 0; nt < 4; ++nt) {
    float bv = bias[nbase + nt * 16 + nlane];
    v8f c;
#pragma unroll
    for (int i = 0; i < 8; ++i) c[i] = bv;
    acc[0][nt] = c; acc[1][nt] = c;
  }

  for (int k0 = 0; k0 < K; k0 += 32) {
    v16bf a0 = load_frag_a(A, mbase,      k0, K, lane);
    v16bf a1 = load_frag_a(A, mbase + 16, k0, K, lane);
#pragma unroll
    for (int nt = 0; nt < 4; ++nt) {
      v16bf b = load_frag_b(BT, nbase + nt * 16, k0, K, lane);
      acc[0][nt] = wmma_bf16(a0, b, acc[0][nt]);
      acc[1][nt] = wmma_bf16(a1, b, acc[1][nt]);
    }
  }

#pragma unroll
  for (int mt = 0; mt < 2; ++mt)
#pragma unroll
    for (int nt = 0; nt < 4; ++nt)
#pragma unroll
      for (int r = 0; r < 8; ++r) {
        int row = mbase + mt * 16 + r + half * 8;
        int col = nbase + nt * 16 + nlane;
        C[(size_t)row * N + col] = acc[mt][nt][r];
      }
}

// ---------------------------------------------------------------------------
// Persistent recurrent scan (one launch per layer).
// Grid = 32 blocks; block jb owns H columns [jb*32, jb*32+32) for all 3 gates.
// Per step: rp = h_bf16[64,1024] @ WrT(slice) via WMMA -> LDS -> gate math ->
// update h (fp32 in d_out state slice, bf16 mirror for next step's A operand)
// -> device-wide barrier (monotonic atomic counter).
// ---------------------------------------------------------------------------
__global__ __launch_bounds__(256) void gru_scan(
    const float* __restrict__ xp,          // [B*T, 3H], row = b*T + t
    const bf16_t* __restrict__ WrT,        // [3H, H]
    const float* __restrict__ rbias,       // [3H]
    const unsigned char* __restrict__ mask,// [B, T] bool
    float* __restrict__ h,                 // [B, H]  (d_out state slice)
    bf16_t* __restrict__ hbf,              // [B, H]
    float* __restrict__ yf,                // [B, T, H] fp32 or null
    bf16_t* __restrict__ ybf,              // [B, T, H] bf16 or null
    unsigned int* __restrict__ cnt, int T) {
  __shared__ float rp[3 * 64 * 32];        // 24 KB gate pre-activations
  const int tid   = threadIdx.x;
  const int lane  = tid & 31;
  const int wave  = tid >> 5;
  const int mt    = wave >> 1;             // 0..3 -> M rows mt*16..+16
  const int nh    = wave & 1;              // which 16-col half of the slice
  const int half  = (lane >> 4) & 1;
  const int nlane = lane & 15;
  const int jb    = blockIdx.x;            // column slice base = jb*32

  for (int t = 0; t < T; ++t) {
    // ---- recurrent GEMM slice: 3 gates x 16 cols x 16 rows per wave ----
    v8f acc[3];
#pragma unroll
    for (int g = 0; g < 3; ++g) {
      float bv = rbias[g * H_DIM + jb * 32 + nh * 16 + nlane];
#pragma unroll
      for (int i = 0; i < 8; ++i) acc[g][i] = bv;
    }
    for (int k0 = 0; k0 < H_DIM; k0 += 32) {
      v16bf a = load_frag_a(hbf, mt * 16, k0, H_DIM, lane);
#pragma unroll
      for (int g = 0; g < 3; ++g) {
        v16bf b = load_frag_b(WrT, g * H_DIM + jb * 32 + nh * 16, k0, H_DIM, lane);
        acc[g] = wmma_bf16(a, b, acc[g]);
      }
    }
#pragma unroll
    for (int g = 0; g < 3; ++g)
#pragma unroll
      for (int r = 0; r < 8; ++r) {
        int row = mt * 16 + r + half * 8;
        int jc  = nh * 16 + nlane;
        rp[(g * 64 + row) * 32 + jc] = acc[g][r];
      }
    __syncthreads();

    // ---- gate math + state update: 64x32 elems, 8 per thread ----
#pragma unroll
    for (int e = 0; e < 8; ++e) {
      int lin = tid + e * 256;
      int b   = lin >> 5;
      int jl  = lin & 31;
      int j   = jb * 32 + jl;
      size_t xrow = (size_t)(b * T + t) * H3;
      float xz = xp[xrow + j];
      float xr = xp[xrow + H_DIM + j];
      float xh = xp[xrow + 2 * H_DIM + j];
      float rz = rp[(0 * 64 + b) * 32 + jl];
      float rr = rp[(1 * 64 + b) * 32 + jl];
      float rh = rp[(2 * 64 + b) * 32 + jl];
      float z  = 1.0f / (1.0f + __expf(-(xz + rz)));
      float r  = 1.0f / (1.0f + __expf(-(xr + rr)));
      float hh = tanhf(xh + r * rh);
      float hp = h[b * H_DIM + j];
      float hn = z * hp + (1.0f - z) * hh;
      hn = mask[b * T + t] ? hn : hp;
      h[b * H_DIM + j]   = hn;
      hbf[b * H_DIM + j] = (bf16_t)hn;
      if (yf)  yf [(size_t)(b * T + t) * H_DIM + j] = hn;
      if (ybf) ybf[(size_t)(b * T + t) * H_DIM + j] = (bf16_t)hn;
    }

    // ---- device-wide barrier: all blocks finish step t before t+1 ----
    __threadfence();
    __syncthreads();
    if (tid == 0) {
      __hip_atomic_fetch_add(cnt, 1u, __ATOMIC_ACQ_REL, __HIP_MEMORY_SCOPE_AGENT);
      unsigned target = (unsigned)gridDim.x * (unsigned)(t + 1);
      while (__hip_atomic_load(cnt, __ATOMIC_ACQUIRE, __HIP_MEMORY_SCOPE_AGENT) < target)
        __builtin_amdgcn_s_sleep(1);
    }
    __syncthreads();
  }
}

// ---------------------------------------------------------------------------
extern "C" void kernel_launch(void* const* d_in, const int* in_sizes, int n_in,
                              void* d_out, int out_size, void* d_ws, size_t ws_size,
                              hipStream_t stream) {
  (void)in_sizes; (void)n_in; (void)out_size; (void)ws_size;
  const float* q_embed = (const float*)d_in[0];
  const float* hidden  = (const float*)d_in[1];
  const unsigned char* mask = (const unsigned char*)d_in[2];
  const float* k0  = (const float*)d_in[3];
  const float* r0  = (const float*)d_in[4];
  const float* b0  = (const float*)d_in[5];
  const float* rb0 = (const float*)d_in[6];
  const float* k1  = (const float*)d_in[7];
  const float* r1  = (const float*)d_in[8];
  const float* b1  = (const float*)d_in[9];
  const float* rb1 = (const float*)d_in[10];

  float* out_now   = (float*)d_out;                                  // [B,T,H]
  float* out_state = out_now + (size_t)B_DIM * T_DIM * H_DIM;        // [2,B,H]

  char* ws = (char*)d_ws;
  size_t off = 0;
  auto alloc = [&](size_t bytes) -> char* {
    char* p = ws + off; off += (bytes + 255) & ~(size_t)255; return p;
  };
  float*   xp   = (float*) alloc((size_t)MTOT * H3 * 4);       // 384 MB, reused
  bf16_t*  xbf  = (bf16_t*)alloc((size_t)MTOT * E_DIM * 2);    // 64 MB
  bf16_t*  y0bf = (bf16_t*)alloc((size_t)MTOT * H_DIM * 2);    // 64 MB
  bf16_t*  wkt0 = (bf16_t*)alloc((size_t)H3 * E_DIM * 2);
  bf16_t*  wrt0 = (bf16_t*)alloc((size_t)H3 * H_DIM * 2);
  bf16_t*  wkt1 = (bf16_t*)alloc((size_t)H3 * H_DIM * 2);
  bf16_t*  wrt1 = (bf16_t*)alloc((size_t)H3 * H_DIM * 2);
  bf16_t*  hbf0 = (bf16_t*)alloc((size_t)B_DIM * H_DIM * 2);
  bf16_t*  hbf1 = (bf16_t*)alloc((size_t)B_DIM * H_DIM * 2);
  unsigned* cnt = (unsigned*)alloc(256);

  init_state<<<(2 * B_DIM * H_DIM + 255) / 256, 256, 0, stream>>>(
      hidden, out_state, hbf0, hbf1, cnt);

  {
    int n = MTOT * E_DIM;
    f32_to_bf16<<<(n / 4 + 255) / 256, 256, 0, stream>>>(q_embed, xbf, n);
  }

  dim3 tb(32, 8);
  transpose_f32_to_bf16<<<dim3(H3 / 32, E_DIM / 32), tb, 0, stream>>>(k0, wkt0, E_DIM, H3);
  transpose_f32_to_bf16<<<dim3(H3 / 32, H_DIM / 32), tb, 0, stream>>>(r0, wrt0, H_DIM, H3);
  transpose_f32_to_bf16<<<dim3(H3 / 32, H_DIM / 32), tb, 0, stream>>>(k1, wkt1, H_DIM, H3);
  transpose_f32_to_bf16<<<dim3(H3 / 32, H_DIM / 32), tb, 0, stream>>>(r1, wrt1, H_DIM, H3);

  dim3 gemm_grid(H3 / 256, MTOT / 64);
  // Layer 0
  gemm_bf16_tn<<<gemm_grid, 256, 0, stream>>>(xbf, wkt0, b0, xp, MTOT, H3, E_DIM);
  gru_scan<<<SCAN_BLOCKS, 256, 0, stream>>>(xp, wrt0, rb0, mask,
                                            out_state, hbf0,
                                            nullptr, y0bf, cnt + 0, T_DIM);
  // Layer 1
  gemm_bf16_tn<<<gemm_grid, 256, 0, stream>>>(y0bf, wkt1, b1, xp, MTOT, H3, H_DIM);
  gru_scan<<<SCAN_BLOCKS, 256, 0, stream>>>(xp, wrt1, rb1, mask,
                                            out_state + B_DIM * H_DIM, hbf1,
                                            out_now, nullptr, cnt + 1, T_DIM);
}
